// DeconvolutionModel_66486093742276
// MI455X (gfx1250) — matrix operand order repaired and checked by reading, not compiled, not run on hardware
//
#include <hip/hip_runtime.h>
#include <hip/hip_bf16.h>

typedef __attribute__((ext_vector_type(16))) __bf16 bf16x16;
typedef __attribute__((ext_vector_type(8)))  __bf16 bf16x8;
typedef __attribute__((ext_vector_type(8)))  float  f32x8;

#define BQ 512
#define MQ 4096
#define CQ 16
#define NCONCQ 50
#define GWAVES 4   // waves per block (16-col tiles)
#define GMT    4   // 16-row tiles per wave

// ---------------------------------------------------------------- prep
__global__ void prep_kernel(const float* __restrict__ X, const float* __restrict__ cov,
                            __bf16* __restrict__ ci, float* __restrict__ Xc) {
    int i = blockIdx.x * blockDim.x + threadIdx.x;          // B*M threads
    if (i >= BQ * MQ) return;
    int b = i >> 12;
    int m = i & (MQ - 1);
    float x = X[i];
    bool valid = !(x != x);                                 // !isnan
    float xc = valid ? x : 0.0f;
    float cv = valid ? cov[i] : 0.0f;
    Xc[i] = xc;
    ci[(size_t)b * (2 * MQ) + m]       = (__bf16)xc;
    ci[(size_t)b * (2 * MQ) + MQ + m]  = (__bf16)cv;
}

// ------------------------------------------------- generic bf16 WMMA GEMM
// out[B x N] = concat_K(A0[B x K0], A1[B x K1]) @ W[K x N] (+ bias if KSPLIT==1)
// All dims compile-time: addresses strength-reduce to immediate offsets.
// KSPLIT>1: grid.z slices K, partial sums written to out[z][B][N] (reduce pass adds bias).
template <int N, int K0, int LDA0, int K1, int LDA1, int KSPLIT>
__global__ __launch_bounds__(32 * GWAVES)
void gemm_kernel(const __bf16* __restrict__ A0, const __bf16* __restrict__ A1,
                 const float* __restrict__ W, const float* __restrict__ bias,
                 float* __restrict__ out) {
    constexpr int K  = K0 + K1;
    constexpr int KC = K / KSPLIT;
    static_assert(KC % 32 == 0, "K chunk must be multiple of 32");
    const int lane = threadIdx.x & 31;
    const int wave = threadIdx.x >> 5;
    const int tn   = blockIdx.x * GWAVES + wave;            // 16-col tile id
    if constexpr (N < 16 * GWAVES) {
        if (tn * 16 >= N) return;                           // uniform per wave
    }
    const int m16  = lane & 15;
    const int half = lane >> 4;
    const int ncol = tn * 16 + m16;
    const int rbase  = blockIdx.y * (16 * GMT);
    const int kstart = blockIdx.z * KC;

    f32x8 acc[GMT] = {};
    const float* Wp = W + (size_t)(kstart + half * 16) * N + ncol;
    for (int kk = 0; kk < KC; kk += 32) {
        const int kb = kstart + kk;
        // B fragment: W[kb + half*16 + j][ncol], j = 0..15 (immediate offsets j*N*4)
        __builtin_prefetch(Wp + (size_t)32 * N, 0, 1);
        bf16x16 bfrag;
#pragma unroll
        for (int j = 0; j < 16; ++j) bfrag[j] = (__bf16)Wp[(size_t)j * N];
        Wp += (size_t)32 * N;
#pragma unroll
        for (int t = 0; t < GMT; ++t) {
            const __bf16* ap;
            if constexpr (K1 == 0) {
                ap = A0 + (size_t)(rbase + t * 16 + m16) * LDA0 + kb + half * 16;
            } else {
                if (kb < K0)
                    ap = A0 + (size_t)(rbase + t * 16 + m16) * LDA0 + kb + half * 16;
                else
                    ap = A1 + (size_t)(rbase + t * 16 + m16) * LDA1 + (kb - K0) + half * 16;
            }
            bf16x8 alo = *(const bf16x8*)ap;                // K = half*16 + 0..7
            bf16x8 ahi = *(const bf16x8*)(ap + 8);          // K = half*16 + 8..15
            bf16x16 afrag = __builtin_shufflevector(alo, ahi,
                0, 1, 2, 3, 4, 5, 6, 7, 8, 9, 10, 11, 12, 13, 14, 15);
            acc[t] = __builtin_amdgcn_wmma_f32_16x16x32_bf16(
                false, afrag, false, bfrag, (short)0, acc[t], false, false);
        }
    }
    if constexpr (KSPLIT == 1) {
        const float bv = bias[ncol];
#pragma unroll
        for (int t = 0; t < GMT; ++t)
#pragma unroll
            for (int r = 0; r < 8; ++r) {
                int orow = rbase + t * 16 + r + half * 8;   // C/D layout
                out[(size_t)orow * N + ncol] = acc[t][r] + bv;
            }
    } else {
        float* po = out + (size_t)blockIdx.z * BQ * N;
#pragma unroll
        for (int t = 0; t < GMT; ++t)
#pragma unroll
            for (int r = 0; r < 8; ++r) {
                int orow = rbase + t * 16 + r + half * 8;
                po[(size_t)orow * N + ncol] = acc[t][r];
            }
    }
}

// sum split-K partials + bias
template <int N, int KSPLIT>
__global__ void reduce_bias_kernel(const float* __restrict__ part,
                                   const float* __restrict__ bias,
                                   float* __restrict__ out) {
    int i = blockIdx.x * blockDim.x + threadIdx.x;
    if (i >= BQ * N) return;
    float s = bias[i % N];
#pragma unroll
    for (int z = 0; z < KSPLIT; ++z) s += part[(size_t)z * BQ * N + i];
    out[i] = s;
}

// ------------------- Wm2 GEMM with fused sigmoid * dist gather * Xc epilogue
__global__ __launch_bounds__(32 * GWAVES)
void gemm_mw_kernel(const __bf16* __restrict__ LH,       // 512 x 256 bf16
                    const float* __restrict__ Wm2,       // 256 x 65536
                    const float* __restrict__ bm2,
                    const float* __restrict__ dist,      // C x NCONC x M
                    const int*   __restrict__ idx,       // B x C
                    const float* __restrict__ Xc,        // B x M
                    float* __restrict__ mw,              // B x C x M (output)
                    __bf16* __restrict__ wi) {           // B x (C*M) bf16
    constexpr int N = CQ * MQ;                           // 65536
    constexpr int K = 256;
    const int lane = threadIdx.x & 31;
    const int wave = threadIdx.x >> 5;
    const int tn   = blockIdx.x * GWAVES + wave;
    const int m16  = lane & 15;
    const int half = lane >> 4;
    const int ncol = tn * 16 + m16;
    const int rbase = blockIdx.y * (16 * GMT);

    f32x8 acc[GMT] = {};
    const float* Wp = Wm2 + (size_t)(half * 16) * N + ncol;
    for (int kb = 0; kb < K; kb += 32) {
        bf16x16 bfrag;
#pragma unroll
        for (int j = 0; j < 16; ++j) bfrag[j] = (__bf16)Wp[(size_t)j * N];
        Wp += (size_t)32 * N;
#pragma unroll
        for (int t = 0; t < GMT; ++t) {
            const __bf16* ap = LH + (size_t)(rbase + t * 16 + m16) * K + kb + half * 16;
            bf16x8 alo = *(const bf16x8*)ap;
            bf16x8 ahi = *(const bf16x8*)(ap + 8);
            bf16x16 afrag = __builtin_shufflevector(alo, ahi,
                0, 1, 2, 3, 4, 5, 6, 7, 8, 9, 10, 11, 12, 13, 14, 15);
            acc[t] = __builtin_amdgcn_wmma_f32_16x16x32_bf16(
                false, afrag, false, bfrag, (short)0, acc[t], false, false);
        }
    }
    const float bv = bm2[ncol];
    const int c    = ncol >> 12;                          // /4096
    const int mcol = ncol & (MQ - 1);
#pragma unroll
    for (int t = 0; t < GMT; ++t)
#pragma unroll
        for (int r = 0; r < 8; ++r) {
            int orow = rbase + t * 16 + r + half * 8;
            float lw = 1.0f / (1.0f + __expf(-(acc[t][r] + bv)));
            int  j   = idx[orow * CQ + c];
            float dw = dist[((size_t)c * NCONCQ + j) * MQ + mcol];
            float mv = lw * dw;
            mw[(size_t)orow * N + ncol] = mv;
            wi[(size_t)orow * N + ncol] = (__bf16)(Xc[(size_t)orow * MQ + mcol] * mv);
        }
}

// ---------------------------------------------------------------- LN + ReLU
__global__ void ln_relu_kernel(const float* __restrict__ pre, const float* __restrict__ g,
                               const float* __restrict__ be, __bf16* __restrict__ out, int N) {
    __shared__ float s1[256];
    __shared__ float s2[256];
    const int row = blockIdx.x;
    const int tid = threadIdx.x;
    const float* x = pre + (size_t)row * N;
    float s = 0.0f, ss = 0.0f;
    for (int i = tid; i < N; i += 256) { float v = x[i]; s += v; ss += v * v; }
    s1[tid] = s; s2[tid] = ss;
    __syncthreads();
    for (int st = 128; st > 0; st >>= 1) {
        if (tid < st) { s1[tid] += s1[tid + st]; s2[tid] += s2[tid + st]; }
        __syncthreads();
    }
    float mean = s1[0] / N;
    float var  = s2[0] / N - mean * mean;
    float inv  = rsqrtf(var + 1e-5f);
    for (int i = tid; i < N; i += 256) {
        float v = (x[i] - mean) * inv * g[i] + be[i];
        out[(size_t)row * N + i] = (__bf16)fmaxf(v, 0.0f);
    }
}

// ---------------------------------------------------------------- small ops
__global__ void sigmoid_kernel(const float* __restrict__ pre, float* __restrict__ out, int n) {
    int i = blockIdx.x * blockDim.x + threadIdx.x;
    if (i < n) out[i] = 1.0f / (1.0f + __expf(-pre[i]));
}

__global__ void argmin_kernel(const float* __restrict__ est, const float* __restrict__ grid,
                              int* __restrict__ idx) {
    int i = blockIdx.x * blockDim.x + threadIdx.x;        // B*C
    if (i >= BQ * CQ) return;
    int c = i & (CQ - 1);
    float e = est[i];
    float best = fabsf(grid[c * NCONCQ] - e);
    int bj = 0;
    for (int j = 1; j < NCONCQ; ++j) {
        float d = fabsf(grid[c * NCONCQ + j] - e);
        if (d < best) { best = d; bj = j; }               // first-min like argmin
    }
    idx[i] = bj;
}

__global__ void softmax_final_kernel(const float* __restrict__ logits,
                                     const float* __restrict__ est,
                                     float* __restrict__ fin) {
    int row = blockIdx.x * blockDim.x + threadIdx.x;
    if (row >= BQ) return;
    const float* l = logits + row * CQ;
    float mx = l[0];
    for (int j = 1; j < CQ; ++j) mx = fmaxf(mx, l[j]);
    float e[CQ], sum = 0.0f;
    for (int j = 0; j < CQ; ++j) { e[j] = __expf(l[j] - mx); sum += e[j]; }
    float inv = 1.0f / sum;
    for (int j = 0; j < CQ; ++j)
        fin[row * CQ + j] = (e[j] * inv + est[row * CQ + j]) * 0.5f;
}

// ---------------------------------------------------------------- launch
extern "C" void kernel_launch(void* const* d_in, const int* in_sizes, int n_in,
                              void* d_out, int out_size, void* d_ws, size_t ws_size,
                              hipStream_t stream) {
    (void)in_sizes; (void)n_in; (void)out_size; (void)ws_size;
    const float* X         = (const float*)d_in[0];
    const float* coverage  = (const float*)d_in[1];
    const float* conc_grid = (const float*)d_in[2];
    const float* dist      = (const float*)d_in[3];
    const float* W1  = (const float*)d_in[4];   const float* b1  = (const float*)d_in[5];
    const float* g1  = (const float*)d_in[6];   const float* be1 = (const float*)d_in[7];
    const float* W2  = (const float*)d_in[8];   const float* b2  = (const float*)d_in[9];
    const float* g2  = (const float*)d_in[10];  const float* be2 = (const float*)d_in[11];
    const float* W3  = (const float*)d_in[12];  const float* b3  = (const float*)d_in[13];
    const float* Wm1 = (const float*)d_in[14];  const float* bm1 = (const float*)d_in[15];
    const float* gm  = (const float*)d_in[16];  const float* bem = (const float*)d_in[17];
    const float* Wm2 = (const float*)d_in[18];  const float* bm2 = (const float*)d_in[19];
    const float* Wf1 = (const float*)d_in[20];  const float* bf1 = (const float*)d_in[21];
    const float* gf1 = (const float*)d_in[22];  const float* bef1= (const float*)d_in[23];
    const float* Wf2 = (const float*)d_in[24];  const float* bf2 = (const float*)d_in[25];
    const float* gf2 = (const float*)d_in[26];  const float* bef2= (const float*)d_in[27];
    const float* Wo  = (const float*)d_in[28];  const float* bo  = (const float*)d_in[29];

    float* outp  = (float*)d_out;
    float* fin   = outp;                 // B*C
    float* est   = outp + BQ * CQ;       // B*C
    float* mw    = outp + 2 * BQ * CQ;   // B*C*M

    char* ws = (char*)d_ws;
    // workspace layout (bytes)
    __bf16* ci     = (__bf16*)(ws + 0);                    //  8.39 MB  B x 2M bf16
    float*  Xc     = (float*) (ws + 8388608);              //  8.39 MB  B x M f32
    float*  pre    = (float*) (ws + 16777216);             //  1.05 MB  B x 512 f32 (reused)
    __bf16* H1     = (__bf16*)(ws + 17825792);             //  0.26 MB
    __bf16* H2     = (__bf16*)(ws + 18087936);             //  0.07 MB
    __bf16* LH     = (__bf16*)(ws + 18153472);             //  0.26 MB
    int*    idx    = (int*)   (ws + 18415616);             //  0.03 MB
    __bf16* F1     = (__bf16*)(ws + 18448384);             //  0.52 MB
    __bf16* F2     = (__bf16*)(ws + 18972672);             //  0.26 MB
    float*  logits = (float*) (ws + 19234816);             //  0.03 MB
    __bf16* wi     = (__bf16*)(ws + 19267584);             // 67.11 MB  B x C*M bf16
    float*  part   = (float*) (ws + 86376448);             //  8.39 MB  split-K partials

    const dim3 gblk(32 * GWAVES);
    const int  rowg = BQ / (16 * GMT);                     // 8

    // input prep
    prep_kernel<<<(BQ * MQ) / 256, 256, 0, stream>>>(X, coverage, ci, Xc);

    // est head: ci@W1 -> LN/ReLU -> @W2 -> LN/ReLU -> @W3 -> sigmoid
    gemm_kernel<256, 2 * MQ, 2 * MQ, 0, 0, 4>
        <<<dim3(4, rowg, 4), gblk, 0, stream>>>(ci, nullptr, W1, nullptr, part);
    reduce_bias_kernel<256, 4><<<(BQ * 256) / 256, 256, 0, stream>>>(part, b1, pre);
    ln_relu_kernel<<<BQ, 256, 0, stream>>>(pre, g1, be1, H1, 256);
    gemm_kernel<64, 256, 256, 0, 0, 1>
        <<<dim3(1, rowg, 1), gblk, 0, stream>>>(H1, nullptr, W2, b2, pre);
    ln_relu_kernel<<<BQ, 256, 0, stream>>>(pre, g2, be2, H2, 64);
    gemm_kernel<16, 64, 64, 0, 0, 1>
        <<<dim3(1, rowg, 1), gblk, 0, stream>>>(H2, nullptr, W3, b3, pre);
    sigmoid_kernel<<<(BQ * CQ + 255) / 256, 256, 0, stream>>>(pre, est, BQ * CQ);
    argmin_kernel<<<(BQ * CQ + 255) / 256, 256, 0, stream>>>(est, conc_grid, idx);

    // mask head: ci@Wm1 -> LN/ReLU -> @Wm2 fused sigmoid*dist*Xc -> mw, wi
    gemm_kernel<256, 2 * MQ, 2 * MQ, 0, 0, 4>
        <<<dim3(4, rowg, 4), gblk, 0, stream>>>(ci, nullptr, Wm1, nullptr, part);
    reduce_bias_kernel<256, 4><<<(BQ * 256) / 256, 256, 0, stream>>>(part, bm1, pre);
    ln_relu_kernel<<<BQ, 256, 0, stream>>>(pre, gm, bem, LH, 256);
    gemm_mw_kernel<<<dim3((CQ * MQ) / 16 / GWAVES, rowg), gblk, 0, stream>>>(
        LH, Wm2, bm2, dist, idx, Xc, mw, wi);

    // fusion head: [wi | cov(bf16 = ci second half)] @ Wf1 -> ... -> softmax
    gemm_kernel<512, CQ * MQ, CQ * MQ, MQ, 2 * MQ, 8>
        <<<dim3(8, rowg, 8), gblk, 0, stream>>>(wi, ci + MQ, Wf1, nullptr, part);
    reduce_bias_kernel<512, 8><<<(BQ * 512) / 256, 256, 0, stream>>>(part, bf1, pre);
    ln_relu_kernel<<<BQ, 256, 0, stream>>>(pre, gf1, bef1, F1, 512);
    gemm_kernel<256, 512, 512, 0, 0, 1>
        <<<dim3(4, rowg, 1), gblk, 0, stream>>>(F1, nullptr, Wf2, bf2, pre);
    ln_relu_kernel<<<BQ, 256, 0, stream>>>(pre, gf2, bef2, F2, 256);
    gemm_kernel<16, 256, 256, 0, 0, 1>
        <<<dim3(1, rowg, 1), gblk, 0, stream>>>(F2, nullptr, Wo, bo, logits);
    softmax_final_kernel<<<(BQ + 255) / 256, 256, 0, stream>>>(logits, est, fin);
}
